// RelationalGCN_85495618995035
// MI455X (gfx1250) — matrix operand order repaired and checked by reading, not compiled, not run on hardware
//
#include <hip/hip_runtime.h>
#include <hip/hip_bf16.h>
#include <stdint.h>

#define N_NODES 50000
#define N_EDGES 800000
#define DIM     128
#define NREL    8
#define NMAT    9        // 8 relations + root
#define NTILES  (N_NODES / 16)   // 3125
#define WF_DW_ROW  4608          // dwords per TDM "row" (16 rows total)
#define WF_TOTAL_DW (NMAT * DIM * DIM / 2)   // 73728 dwords = 288KB

typedef __attribute__((ext_vector_type(16))) __bf16 v16bf;
typedef __attribute__((ext_vector_type(8)))  float  v8f;
typedef unsigned int v4u __attribute__((ext_vector_type(4)));
typedef int v8i __attribute__((ext_vector_type(8)));
typedef int v4i __attribute__((ext_vector_type(4)));

union BF16x16 { uint32_t u[8]; v16bf v; };
union BFPair  { uint32_t u; __bf16 h[2]; };
union BF4     { uint32_t u[2]; __bf16 h[4]; };

// ---------------- small utility kernels ----------------

__global__ void __launch_bounds__(256) zero_cnt_k(int* __restrict__ cnt) {
    int i = blockIdx.x * 256 + threadIdx.x;
    if (i < N_NODES * NREL) cnt[i] = 0;
}

__global__ void __launch_bounds__(256) count_edges_k(const int* __restrict__ eidx,
                                                     const int* __restrict__ etype,
                                                     int* __restrict__ cnt) {
    int e = blockIdx.x * 256 + threadIdx.x;
    if (e >= N_EDGES) return;
    int d = eidx[N_EDGES + e];
    int t = etype[e];
    atomicAdd(&cnt[(size_t)d * NREL + t], 1);
}

__global__ void __launch_bounds__(256) make_inv_k(const int* __restrict__ cnt,
                                                  float* __restrict__ inv) {
    int i = blockIdx.x * 256 + threadIdx.x;
    if (i < N_NODES * NREL) {
        int c = cnt[i];
        inv[i] = 1.0f / (float)(c > 0 ? c : 1);
    }
}

// Pack f32 weights -> bf16, laid out so a B fragment is 8 CONTIGUOUS dwords:
// dword index = (((mat*4 + kt)*2 + half)*128 + n)*8 + v , pair (k,k+1) within dword.
// k = kt*32 + half*16 + v*2 + par
__global__ void __launch_bounds__(256) pack_weights_k(const float* __restrict__ W,
                                                      const float* __restrict__ root,
                                                      __bf16* __restrict__ wf) {
    int i = blockIdx.x * 256 + threadIdx.x;
    if (i >= NMAT * DIM * DIM) return;
    int mat = i / (DIM * DIM);
    int rem = i % (DIM * DIM);
    int k = rem / DIM, n = rem % DIM;
    float val = (mat < NREL) ? W[(size_t)mat * DIM * DIM + rem] : root[rem];
    int kt = k >> 5, half = (k >> 4) & 1, v = (k >> 1) & 7, par = k & 1;
    size_t dw = ((((size_t)mat * 4 + kt) * 2 + half) * DIM + n) * 8 + v;
    wf[dw * 2 + par] = (__bf16)val;
}

__global__ void __launch_bounds__(256) relu_k(float* __restrict__ h) {
    int i = blockIdx.x * 256 + threadIdx.x;
    if (i < N_NODES * DIM) h[i] = fmaxf(h[i], 0.0f);
}

// ---------------- persistent WMMA GEMM ----------------
// 256 blocks x 9 waves. All 9 weight matrices (288KB bf16) TDM-loaded into LDS once
// per block; grid-stride over 16-node tiles. Wave w computes relation w, wave 8 root+bias.
__global__ void __launch_bounds__(288)
rgcn_gemm_k(const float* __restrict__ act,     // [N, D] f32
            const __bf16* __restrict__ wf,     // packed bf16 weights (see pack_weights_k)
            const float* __restrict__ bias,    // [D]
            __bf16* __restrict__ Y,            // [R, N, D] bf16
            float* __restrict__ out)           // [N, D] f32 (root + bias)
{
    __shared__ __bf16 ldsW[NMAT * DIM * DIM];  // 294,912 B
    __shared__ __bf16 tile[16][136];           // 4,352 B (padded pitch)

    const int tid  = threadIdx.x;
    const int wid  = tid >> 5;
    const int lane = tid & 31;
    const int half = lane >> 4;
    const int m    = lane & 15;

    // --- TDM: bulk-copy all packed weights global -> LDS (wave 0 issues) ---
    if (wid == 0) {
        uint64_t ga = (uint64_t)(uintptr_t)wf;
        uint32_t lb = (uint32_t)(uintptr_t)&ldsW[0];
        // D# group0: count=1 | lds_addr | global_addr | type=2
        v4u g0 = { 1u, lb, (uint32_t)ga,
                   (uint32_t)((ga >> 32) & 0x01FFFFFFu) | 0x80000000u };
        // D# group1: data_size=4B; tensor 4608x16 dwords; tile 4608x16; strides 4608
        v8i g1 = { (int)(2u << 16),                       // data_size=2 (4B)
                   (int)((WF_DW_ROW & 0xFFFF) << 16),     // tensor_dim0 lo
                   (int)(16 << 16),                       // dim0 hi=0 | tensor_dim1=16
                   (int)((WF_DW_ROW & 0xFFFF) << 16),     // dim1 hi=0 | tile_dim0
                   16,                                    // tile_dim1=16, tile_dim2=0
                   (int)WF_DW_ROW,                        // tensor_dim0_stride lo
                   (int)((WF_DW_ROW & 0xFFFF) << 16),     // stride0 hi=0 | stride1 lo
                   0 };                                   // stride1 hi
        v4i g2 = { 1, 1, (int)WF_TOTAL_DW, 0 };           // dim2=1, dim3=1, stride2
        v4i g3 = { (int)WF_TOTAL_DW, (1 << 16), 0, 0 };   // stride3, dim4=1, tile4=0
        v8i g4 = { 0, 0, 0, 0, 0, 0, 0, 0 };              // unused trailing group
        __builtin_amdgcn_tensor_load_to_lds(g0, g1, g2, g3, g4, 0);
        __builtin_amdgcn_s_wait_tensorcnt(0);
    }
    __syncthreads();

    const uint32_t* ldsWdw = (const uint32_t*)&ldsW[0];
    const uint32_t* arow   = (const uint32_t*)&tile[m][0];

    for (int t = blockIdx.x; t < NTILES; t += gridDim.x) {
        const int m0 = t * 16;
        // stage activation tile (f32 -> bf16), float4 loads
        for (int i = tid; i < 512; i += 288) {
            int r = i >> 5, c4 = i & 31;
            float4 f = ((const float4*)(act + (size_t)(m0 + r) * DIM))[c4];
            BF4 p;
            p.h[0] = (__bf16)f.x; p.h[1] = (__bf16)f.y;
            p.h[2] = (__bf16)f.z; p.h[3] = (__bf16)f.w;
            uint32_t* dst = (uint32_t*)&tile[r][c4 * 4];
            dst[0] = p.u[0]; dst[1] = p.u[1];
        }
        __syncthreads();

        v8f acc[8] = {};
#pragma unroll
        for (int kt = 0; kt < 4; ++kt) {
            const int kb2 = kt * 16;
            BF16x16 a;
            // A 16x32 bf16, this code's k order (B packed alike): lanes<16 K=0..15 ; lanes>=16 K=16..31
#pragma unroll
            for (int v = 0; v < 8; ++v) {
                int dw = kb2 + 8 * half + v;
                a.u[v] = arow[dw];
            }
#pragma unroll
            for (int nt = 0; nt < 8; ++nt) {
                const uint32_t* bw =
                    ldsWdw + ((size_t)(((wid * 4 + kt) * 2 + half) * DIM + nt * 16 + m)) * 8;
                BF16x16 b;      // 8 contiguous dwords -> 2x ds_load_b128
#pragma unroll
                for (int v = 0; v < 8; ++v) b.u[v] = bw[v];
                acc[nt] = __builtin_amdgcn_wmma_f32_16x16x32_bf16(
                    false, a.v, false, b.v, (short)0, acc[nt], false, false);
            }
        }

        // C/D: lanes<16 -> M=v, N=lane ; lanes>=16 -> M=v+8, N=lane-16
        if (wid < NREL) {
#pragma unroll
            for (int nt = 0; nt < 8; ++nt)
#pragma unroll
                for (int v = 0; v < 8; ++v) {
                    int row = v + 8 * half;
                    Y[((size_t)wid * N_NODES + (m0 + row)) * DIM + nt * 16 + m] =
                        (__bf16)acc[nt][v];
                }
        } else {
#pragma unroll
            for (int nt = 0; nt < 8; ++nt)
#pragma unroll
                for (int v = 0; v < 8; ++v) {
                    int row = v + 8 * half;
                    int col = nt * 16 + m;
                    out[(size_t)(m0 + row) * DIM + col] = acc[nt][v] + bias[col];
                }
        }
        __syncthreads();   // tile reuse across grid-stride iterations
    }
}

// ---------------- edge scatter: out[dst] += Y[t][src] * inv[dst,t] ----------------
// 4 edges per wave; prefetch Y rows before the atomic burst.
#define EPW 4
__global__ void __launch_bounds__(256)
edge_pass_k(const int* __restrict__ eidx, const int* __restrict__ etype,
            const __bf16* __restrict__ Y, const float* __restrict__ inv,
            float* __restrict__ out)
{
    int w    = (blockIdx.x * 256 + threadIdx.x) >> 5;
    int lane = threadIdx.x & 31;
    int e0   = w * EPW;                     // grid sized exactly: no bounds check
    int s[EPW], d[EPW], tt[EPW];
    const uint2* rows[EPW];
#pragma unroll
    for (int j = 0; j < EPW; ++j) {
        int e = e0 + j;
        s[j]  = eidx[e];
        d[j]  = eidx[N_EDGES + e];
        tt[j] = etype[e];
        rows[j] = (const uint2*)(Y + ((size_t)tt[j] * N_NODES + s[j]) * DIM);
        __builtin_prefetch((const void*)(rows[j] + lane), 0, 0);   // global_prefetch_b8
    }
#pragma unroll
    for (int j = 0; j < EPW; ++j) {
        float sc = inv[(size_t)d[j] * NREL + tt[j]];
        uint2 p = rows[j][lane];
        BFPair lo, hi; lo.u = p.x; hi.u = p.y;
        float* o = out + (size_t)d[j] * DIM + lane * 4;
        unsafeAtomicAdd(o + 0, (float)lo.h[0] * sc);
        unsafeAtomicAdd(o + 1, (float)lo.h[1] * sc);
        unsafeAtomicAdd(o + 2, (float)hi.h[0] * sc);
        unsafeAtomicAdd(o + 3, (float)hi.h[1] * sc);
    }
}

// ---------------- row L2-normalize ----------------
__global__ void __launch_bounds__(256)
normalize_k(float* __restrict__ out)
{
    int w    = (blockIdx.x * 256 + threadIdx.x) >> 5;
    int lane = threadIdx.x & 31;
    if (w >= N_NODES) return;
    float4* row = (float4*)(out + (size_t)w * DIM);
    float4 v = row[lane];
    float ss = v.x * v.x + v.y * v.y + v.z * v.z + v.w * v.w;
#pragma unroll
    for (int off = 16; off > 0; off >>= 1) ss += __shfl_xor(ss, off, 32);
    float s = 1.0f / fmaxf(sqrtf(ss), 1e-12f);
    v.x *= s; v.y *= s; v.z *= s; v.w *= s;
    row[lane] = v;
}

// ---------------- host launcher ----------------
extern "C" void kernel_launch(void* const* d_in, const int* in_sizes, int n_in,
                              void* d_out, int out_size, void* d_ws, size_t ws_size,
                              hipStream_t stream) {
    const float* x     = (const float*)d_in[0];
    const int*   eidx  = (const int*)  d_in[1];   // [2, E]
    const int*   etype = (const int*)  d_in[2];   // [E]
    const float* W1    = (const float*)d_in[3];
    const float* root1 = (const float*)d_in[4];
    const float* b1    = (const float*)d_in[5];
    const float* W2    = (const float*)d_in[6];
    const float* root2 = (const float*)d_in[7];
    const float* b2    = (const float*)d_in[8];
    float* outp = (float*)d_out;

    char* ws = (char*)d_ws;
    float*  h   = (float*)  (ws);                  //  25,600,000 B  [N,D] f32
    __bf16* Y   = (__bf16*) (ws + 25600000);       // 102,400,000 B  [R,N,D] bf16
    __bf16* wf1 = (__bf16*) (ws + 128000000);      //     294,912 B
    __bf16* wf2 = (__bf16*) (ws + 128294912);      //     294,912 B
    int*    cnt = (int*)    (ws + 128589824);      //   1,600,000 B
    float*  inv = (float*)  (ws + 130189824);      //   1,600,000 B

    // per-(dst, relation) degree counts (shared by both layers)
    zero_cnt_k   <<<(N_NODES * NREL + 255) / 256, 256, 0, stream>>>(cnt);
    count_edges_k<<<(N_EDGES + 255) / 256, 256, 0, stream>>>(eidx, etype, cnt);
    make_inv_k   <<<(N_NODES * NREL + 255) / 256, 256, 0, stream>>>(cnt, inv);

    // pack weights to bf16 (B-fragment-contiguous layout)
    pack_weights_k<<<(NMAT * DIM * DIM + 255) / 256, 256, 0, stream>>>(W1, root1, wf1);
    pack_weights_k<<<(NMAT * DIM * DIM + 255) / 256, 256, 0, stream>>>(W2, root2, wf2);

    // layer 1
    rgcn_gemm_k<<<256, 288, 0, stream>>>(x, wf1, b1, Y, h);
    edge_pass_k<<<N_EDGES / (EPW * 8), 256, 0, stream>>>(eidx, etype, Y, inv, h);
    relu_k     <<<(N_NODES * DIM + 255) / 256, 256, 0, stream>>>(h);

    // layer 2
    rgcn_gemm_k<<<256, 288, 0, stream>>>(h, wf2, b2, Y, outp);
    edge_pass_k<<<N_EDGES / (EPW * 8), 256, 0, stream>>>(eidx, etype, Y, inv, outp);

    normalize_k<<<(N_NODES + 7) / 8, 256, 0, stream>>>(outp);
}
